// LlamaAttention_14293651161373
// MI455X (gfx1250) — matrix operand order repaired and checked by reading, not compiled, not run on hardware
//
#include <hip/hip_runtime.h>
#include <hip/hip_bf16.h>

typedef __attribute__((ext_vector_type(16))) _Float16 v16h;
typedef __attribute__((ext_vector_type(8)))  _Float16 v8h;
typedef __attribute__((ext_vector_type(8)))  float    v8f;

#define BB 2
#define SS 2048
#define DD 1024
#define HH 16
#define HD 64

static __device__ __forceinline__ v16h make_v16(v8h lo, v8h hi) {
    v16h r;
#pragma unroll
    for (int i = 0; i < 8; ++i) { r[i] = lo[i]; r[i + 8] = hi[i]; }
    return r;
}

static __device__ __forceinline__ float half_reduce_max(float v) {
    v = fmaxf(v, __shfl_xor(v, 1, 32));
    v = fmaxf(v, __shfl_xor(v, 2, 32));
    v = fmaxf(v, __shfl_xor(v, 4, 32));
    v = fmaxf(v, __shfl_xor(v, 8, 32));
    return v;
}
static __device__ __forceinline__ float half_reduce_sum(float v) {
    v += __shfl_xor(v, 1, 32);
    v += __shfl_xor(v, 2, 32);
    v += __shfl_xor(v, 4, 32);
    v += __shfl_xor(v, 8, 32);
    return v;
}

// Async global -> LDS copy of 16 bytes (per-lane addresses), ASYNCcnt-tracked.
// VDST operand = LDS byte address (low 32 bits of generic shared pointer),
// VADDR = 64-bit global address.
static __device__ __forceinline__ void async_copy_b128(void* lds_ptr, const void* gptr) {
    unsigned lds_lo = (unsigned)(unsigned long long)(uintptr_t)lds_ptr;
    unsigned long long ga = (unsigned long long)(uintptr_t)gptr;
    asm volatile("global_load_async_to_lds_b128 %0, %1, off"
                 :: "v"(lds_lo), "v"(ga) : "memory");
}
static __device__ __forceinline__ void wait_async0() {
    asm volatile("s_wait_asynccnt 0" ::: "memory");
}

// ---------------------------------------------------------------------------
// f32 -> f16 convert
// ---------------------------------------------------------------------------
__global__ void cvt_f32_f16(const float* __restrict__ src, _Float16* __restrict__ dst, int n) {
    for (int i = blockIdx.x * blockDim.x + threadIdx.x; i < n; i += gridDim.x * blockDim.x)
        dst[i] = (_Float16)src[i];
}

// ---------------------------------------------------------------------------
// WMMA GEMM:  C[M,N] = A[M,K](f16,rm) * Bw[K,N](f16,rm) + bias[N]
// MODE 0: write f32 row-major to out32
// MODE 1: write f16 head-split [B,H,S,HD] to out16 (assumes M=B*S, N=D)
// Block tile 128x128, 8 waves (2x4), wave tile 64x32, K-step 32.
// A tile staged via global_load_async_to_lds_b128 (ASYNCcnt); B staged through
// VGPRs (transpose in flight).
// ---------------------------------------------------------------------------
template <int MODE>
__global__ __launch_bounds__(256) void gemm_f16_wmma(
    const _Float16* __restrict__ A, const _Float16* __restrict__ Bw,
    const float* __restrict__ bias, _Float16* __restrict__ out16,
    float* __restrict__ out32, int M, int N, int K)
{
    __shared__ _Float16 Ah[128 * 40];   // [row][k], pitch 40
    __shared__ _Float16 Bt[128 * 40];   // [n][k],  pitch 40 (B transposed)

    const int t    = threadIdx.x;
    const int lane = t & 31;
    const int wave = t >> 5;       // 0..7
    const int wm   = wave >> 2;    // 0..1
    const int wn   = wave & 3;     // 0..3
    const int bm   = blockIdx.x * 128;
    const int bn   = blockIdx.y * 128;
    const int g    = lane >> 4;    // half-wave group
    const int l16  = lane & 15;

    v8f acc[4][2];
#pragma unroll
    for (int i = 0; i < 4; ++i)
#pragma unroll
        for (int j = 0; j < 2; ++j)
#pragma unroll
            for (int r = 0; r < 8; ++r) acc[i][j][r] = 0.0f;

    for (int k0 = 0; k0 < K; k0 += 32) {
        // ---- stage A tile (128x32) asynchronously: 2x b128 per thread ----
#pragma unroll
        for (int i = 0; i < 2; ++i) {
            int row = i * 64 + (t >> 2);
            int c8  = (t & 3) * 8;
            async_copy_b128(Ah + row * 40 + c8,
                            A + (size_t)(bm + row) * K + k0 + c8);
        }
        // ---- stage B tile transposed (32x128 -> Bt[n][k]) ----
#pragma unroll
        for (int i = 0; i < 2; ++i) {
            int kk = i * 16 + (t >> 4);
            int n8 = (t & 15) * 8;
            v8h v = *(const v8h*)(Bw + (size_t)(k0 + kk) * N + bn + n8);
#pragma unroll
            for (int j = 0; j < 8; ++j) Bt[(n8 + j) * 40 + kk] = v[j];
        }
        // prefetch next K-step's B rows into cache while we compute
        if (k0 + 32 < K) {
            int kk = (t >> 4);
            int n8 = (t & 15) * 8;
            __builtin_prefetch(Bw + (size_t)(k0 + 32 + kk) * N + bn + n8, 0, 1);
            __builtin_prefetch(Bw + (size_t)(k0 + 48 + kk) * N + bn + n8, 0, 1);
        }
        wait_async0();
        __syncthreads();

        // ---- fragment loads ----
        v16h aF[4], bF[2];
#pragma unroll
        for (int fm = 0; fm < 4; ++fm) {
            int m_l = wm * 64 + fm * 16 + l16;
            v8h lo = *(const v8h*)(Ah + m_l * 40 + g * 8);
            v8h hi = *(const v8h*)(Ah + m_l * 40 + 16 + g * 8);
            aF[fm] = make_v16(lo, hi);
        }
#pragma unroll
        for (int fn = 0; fn < 2; ++fn) {
            int n_l = wn * 32 + fn * 16 + l16;
            v8h lo = *(const v8h*)(Bt + n_l * 40 + g * 8);
            v8h hi = *(const v8h*)(Bt + n_l * 40 + 16 + g * 8);
            bF[fn] = make_v16(lo, hi);
        }
        // ---- 8 WMMAs ----
#pragma unroll
        for (int fm = 0; fm < 4; ++fm)
#pragma unroll
            for (int fn = 0; fn < 2; ++fn)
                acc[fm][fn] = __builtin_amdgcn_wmma_f32_16x16x32_f16(
                    false, aF[fm], false, bF[fn], (short)0, acc[fm][fn], false, false);
        __syncthreads();
    }

    // ---- epilogue (straight-line per MODE) ----
#pragma unroll
    for (int fm = 0; fm < 4; ++fm)
#pragma unroll
        for (int fn = 0; fn < 2; ++fn) {
            const int col = bn + wn * 32 + fn * 16 + l16;
            const float bv = bias[col];
            const int row0 = bm + wm * 64 + fm * 16 + g * 8;
            if (MODE == 0) {
                float* p = out32 + (size_t)row0 * N + col;
#pragma unroll
                for (int r = 0; r < 8; ++r)
                    p[(size_t)r * N] = acc[fm][fn][r] + bv;
            } else {
                const int b  = row0 >> 11;          // / SS
                const int s0 = row0 & (SS - 1);
                const int h  = col >> 6;            // / HD
                const int hd = col & (HD - 1);
                _Float16* p = out16 + ((((size_t)b * HH + h) * SS + s0) << 6) + hd;
#pragma unroll
                for (int r = 0; r < 8; ++r)
                    p[(size_t)r << 6] = (_Float16)(acc[fm][fn][r] + bv);
            }
        }
}

// ---------------------------------------------------------------------------
// Flash-style attention over f16 Q,K,V in [B,H,S,HD] layout.
// Block = (b,h) x 128-query tile; 8 waves x 16 queries; key tiles of 32.
// ---------------------------------------------------------------------------
__global__ __launch_bounds__(256) void attn_wmma(
    const _Float16* __restrict__ Qh, const _Float16* __restrict__ Kh,
    const _Float16* __restrict__ Vh, const float* __restrict__ amask,
    _Float16* __restrict__ Ctx)
{
    __shared__ _Float16 Vt[HD * 40];        // V tile transposed: [hd][key]
    __shared__ _Float16 Pb[8 * 16 * 40];    // per-wave P tile: [m][k], pitch 40

    const int t    = threadIdx.x;
    const int lane = t & 31;
    const int wave = t >> 5;
    const int g    = lane >> 4;
    const int l16  = lane & 15;
    const int bh   = blockIdx.x;            // 0..31
    const int b    = bh >> 4;
    const int h    = bh & 15;
    const int qt   = blockIdx.y;            // 0..15

    const _Float16* Qb = Qh + (size_t)bh * SS * HD;
    const _Float16* Kb = Kh + (size_t)bh * SS * HD;
    const _Float16* Vb = Vh + (size_t)bh * SS * HD;
    const float*    mr = amask + (size_t)b * SS;

    // Q A-fragments (16 queries x 64 hd -> 2 k-steps of 32)
    const int qrow = qt * 128 + wave * 16 + l16;
    v16h qF[2];
#pragma unroll
    for (int ks = 0; ks < 2; ++ks) {
        v8h lo = *(const v8h*)(Qb + (size_t)qrow * HD + ks * 32 + g * 8);
        v8h hi = *(const v8h*)(Qb + (size_t)qrow * HD + ks * 32 + 16 + g * 8);
        qF[ks] = make_v16(lo, hi);
    }

    float mrun[8], lrun[8];
    v8f o[4];
#pragma unroll
    for (int r = 0; r < 8; ++r) { mrun[r] = -1e30f; lrun[r] = 0.0f; }
#pragma unroll
    for (int f = 0; f < 4; ++f)
#pragma unroll
        for (int r = 0; r < 8; ++r) o[f][r] = 0.0f;

    for (int kt = 0; kt < SS; kt += 32) {
        // stage V tile transposed (32 keys x 64 hd)
        {
            int key = t >> 3;
            int hd8 = (t & 7) * 8;
            v8h vv = *(const v8h*)(Vb + (size_t)(kt + key) * HD + hd8);
#pragma unroll
            for (int j = 0; j < 8; ++j) Vt[(hd8 + j) * 40 + key] = vv[j];
        }
        // prefetch next key tile of K and V
        if (kt + 32 < SS) {
            int key = t >> 3;
            int hd8 = (t & 7) * 8;
            __builtin_prefetch(Kb + (size_t)(kt + 32 + key) * HD + hd8, 0, 1);
            __builtin_prefetch(Vb + (size_t)(kt + 32 + key) * HD + hd8, 0, 1);
        }
        __syncthreads();

        // scores for 16 queries x 32 keys
        v8f s[2];
#pragma unroll
        for (int nt = 0; nt < 2; ++nt) {
#pragma unroll
            for (int r = 0; r < 8; ++r) s[nt][r] = 0.0f;
            int key = kt + nt * 16 + l16;
#pragma unroll
            for (int ks = 0; ks < 2; ++ks) {
                v8h lo = *(const v8h*)(Kb + (size_t)key * HD + ks * 32 + g * 8);
                v8h hi = *(const v8h*)(Kb + (size_t)key * HD + ks * 32 + 16 + g * 8);
                v16h kF = make_v16(lo, hi);
                s[nt] = __builtin_amdgcn_wmma_f32_16x16x32_f16(
                    false, qF[ks], false, kF, (short)0, s[nt], false, false);
            }
            float madd = (1.0f - mr[key]) * -10000.0f;
#pragma unroll
            for (int r = 0; r < 8; ++r) s[nt][r] = s[nt][r] * 0.125f + madd;
        }

        // online softmax (per-row state; rows live across 16-lane halves)
        float alpha[8];
#pragma unroll
        for (int r = 0; r < 8; ++r) {
            float mx = half_reduce_max(fmaxf(s[0][r], s[1][r]));
            float mnew = fmaxf(mrun[r], mx);
            alpha[r] = __expf(mrun[r] - mnew);
            mrun[r] = mnew;
            s[0][r] = __expf(s[0][r] - mnew);
            s[1][r] = __expf(s[1][r] - mnew);
            float sm = half_reduce_sum(s[0][r] + s[1][r]);
            lrun[r] = lrun[r] * alpha[r] + sm;
        }

        // transpose P (C-layout) -> A-layout via wave-private LDS
        _Float16* Pw = Pb + wave * 16 * 40;
#pragma unroll
        for (int r = 0; r < 8; ++r) {
            int m = r + g * 8;
            Pw[m * 40 + l16]      = (_Float16)s[0][r];
            Pw[m * 40 + 16 + l16] = (_Float16)s[1][r];
        }
        asm volatile("s_wait_dscnt 0" ::: "memory");
        v8h plo = *(const v8h*)(Pw + l16 * 40 + g * 8);
        v8h phi = *(const v8h*)(Pw + l16 * 40 + 16 + g * 8);
        v16h pF = make_v16(plo, phi);

        // rescale and accumulate O += P * V
#pragma unroll
        for (int f = 0; f < 4; ++f) {
#pragma unroll
            for (int r = 0; r < 8; ++r) o[f][r] *= alpha[r];
            v8h lo = *(const v8h*)(Vt + (f * 16 + l16) * 40 + g * 8);
            v8h hi = *(const v8h*)(Vt + (f * 16 + l16) * 40 + 16 + g * 8);
            v16h vF = make_v16(lo, hi);
            o[f] = __builtin_amdgcn_wmma_f32_16x16x32_f16(
                false, pF, false, vF, (short)0, o[f], false, false);
        }
        __syncthreads();
    }

    // normalize and write context in [B,S,D] f16 layout
#pragma unroll
    for (int f = 0; f < 4; ++f)
#pragma unroll
        for (int r = 0; r < 8; ++r) {
            int row = qt * 128 + wave * 16 + r + g * 8;
            int col = h * HD + f * 16 + l16;
            float val = o[f][r] / lrun[r];
            Ctx[((size_t)(b * SS + row)) * DD + col] = (_Float16)val;
        }
}

// ---------------------------------------------------------------------------
extern "C" void kernel_launch(void* const* d_in, const int* in_sizes, int n_in,
                              void* d_out, int out_size, void* d_ws, size_t ws_size,
                              hipStream_t stream) {
    const float* hs    = (const float*)d_in[0];
    const float* amask = (const float*)d_in[1];
    const float* q_w   = (const float*)d_in[2];
    const float* q_b   = (const float*)d_in[3];
    const float* k_w   = (const float*)d_in[4];
    const float* k_b   = (const float*)d_in[5];
    const float* v_w   = (const float*)d_in[6];
    const float* v_b   = (const float*)d_in[7];
    const float* o_w   = (const float*)d_in[8];
    const float* o_b   = (const float*)d_in[9];

    const int M = BB * SS;         // 4096
    const int NX = M * DD;         // 4,194,304
    const int NW = DD * DD;        // 1,048,576

    _Float16* Xh = (_Float16*)d_ws;
    _Float16* Wq = Xh + NX;
    _Float16* Wk = Wq + NW;
    _Float16* Wv = Wk + NW;
    _Float16* Wo = Wv + NW;
    _Float16* Qh = Wo + NW;
    _Float16* Kh = Qh + NX;
    _Float16* Vh = Kh + NX;
    _Float16* Ch = Vh + NX;

    cvt_f32_f16<<<2048, 256, 0, stream>>>(hs,  Xh, NX);
    cvt_f32_f16<<<1024, 256, 0, stream>>>(q_w, Wq, NW);
    cvt_f32_f16<<<1024, 256, 0, stream>>>(k_w, Wk, NW);
    cvt_f32_f16<<<1024, 256, 0, stream>>>(v_w, Wv, NW);
    cvt_f32_f16<<<1024, 256, 0, stream>>>(o_w, Wo, NW);

    dim3 gg(M / 128, DD / 128);    // 32 x 8
    gemm_f16_wmma<1><<<gg, 256, 0, stream>>>(Xh, Wq, q_b, Qh, nullptr, M, DD, DD);
    gemm_f16_wmma<1><<<gg, 256, 0, stream>>>(Xh, Wk, k_b, Kh, nullptr, M, DD, DD);
    gemm_f16_wmma<1><<<gg, 256, 0, stream>>>(Xh, Wv, v_b, Vh, nullptr, M, DD, DD);

    dim3 ga(BB * HH, SS / 128);    // 32 x 16
    attn_wmma<<<ga, 256, 0, stream>>>(Qh, Kh, Vh, amask, Ch);

    gemm_f16_wmma<0><<<gg, 256, 0, stream>>>(Ch, Wo, o_b, nullptr, (float*)d_out, M, DD, DD);
}
